// BertCRFDecoder_51221779972192
// MI455X (gfx1250) — matrix operand order repaired and checked by reading, not compiled, not run on hardware
//
#include <hip/hip_runtime.h>
#include <hip/hip_bf16.h>
#include <stdint.h>

typedef __attribute__((ext_vector_type(16))) _Float16 v16h;
typedef __attribute__((ext_vector_type(8)))  _Float16 v8h;
typedef __attribute__((ext_vector_type(8)))  float    v8f;

#define Bc 128
#define Sc 512
#define Tc 256
#define FB 16      // batches per forward block (WMMA M dim)
#define AP 264     // padded f16 row stride for A staging (bank-conflict avoidance)
#define BP 264     // padded f16 row stride for B (exp(trans)^T)

// ---------------------------------------------------------------------------
// Forward algorithm (denominator) in the exp domain via WMMA f16 GEMM.
// One block handles FB=16 batches. State per step: score[16][256] (f32, LDS),
// per-batch running max mmax[16], and a = exp(score - mmax) in f16 (A matrix).
// nxt[b][j] = mmax[b] + log( (a @ exp(trans))[b][j] ) + emit[b][t][j]
// ---------------------------------------------------------------------------
__global__ void crf_forward_kernel(const float* __restrict__ logits,
                                   const unsigned char* __restrict__ mask,
                                   const float* __restrict__ start_t,
                                   const float* __restrict__ end_t,
                                   const float* __restrict__ trans,
                                   float* __restrict__ den) {
  extern __shared__ char smem[];
  float*    score   = (float*)smem;                 // FB*Tc          (16384 B)
  float*    emit_sh = score + FB * Tc;              // FB*Tc          (16384 B)
  float*    end_sh  = emit_sh + FB * Tc;            // Tc             ( 1024 B)
  float*    red     = end_sh + Tc;                  // 256            ( 1024 B)
  float*    mmax    = red + 256;                    // FB             (   64 B)
  int*      len_sh  = (int*)(mmax + FB);            // FB             (   64 B)
  _Float16* aexp    = (_Float16*)(len_sh + FB);     // FB*AP          ( 8448 B)
  _Float16* bT      = aexp + FB * AP;               // Tc*BP          (135168 B)

  const int tid = threadIdx.x;          // 256 threads = 8 waves
  const int b0  = blockIdx.x * FB;

  // B matrix, stored transposed: bT[n][k] = exp(trans[k][n]) (time invariant)
  for (int k = 0; k < Tc; ++k)
    bT[tid * BP + k] = (_Float16)__expf(trans[k * Tc + tid]);
  end_sh[tid] = end_t[tid];

  // sequence lengths for this block's 16 batches
  { int bb = tid >> 4, rl = tid & 15; int c = 0;
    for (int t = rl; t < Sc; t += 16) c += (int)mask[(size_t)(b0 + bb) * Sc + t];
    red[tid] = (float)c; }
  __syncthreads();
  if (tid < FB) { int c = 0;
    for (int s = 0; s < 16; ++s) c += (int)red[tid * 16 + s];
    len_sh[tid] = c; }
  __syncthreads();

  // score_0 = start + logits[:,0,:]
  for (int idx = tid; idx < FB * Tc; idx += 256) {
    int bb = idx >> 8, j = idx & 255;
    score[bb * Tc + j] = start_t[j] + logits[((size_t)(b0 + bb) * Sc) * Tc + j];
  }
  __syncthreads();

  // recompute running max and f16 A matrix from score
  auto recompute = [&]() {
    int rb = tid >> 4, rl = tid & 15;
    float m = -3.0e38f;
    for (int s = 0; s < 16; ++s) m = fmaxf(m, score[rb * Tc + rl + 16 * s]);
    red[tid] = m;
    __syncthreads();
    if (tid < FB) { float mm = -3.0e38f;
      for (int s = 0; s < 16; ++s) mm = fmaxf(mm, red[tid * 16 + s]);
      mmax[tid] = mm; }
    __syncthreads();
    for (int idx = tid; idx < FB * Tc; idx += 256) {
      int bb = idx >> 8, k = idx & 255;
      aexp[bb * AP + k] = (_Float16)__expf(score[bb * Tc + k] - mmax[bb]);
    }
    __syncthreads();
  };
  recompute();

  // WMMA lane mapping (wave32): A: m=L&15, K half h=L>>4. B/D: n=L&15.
  const int w = tid >> 5, L = tid & 31;
  const int mrow = L & 15, hh = L >> 4, ncol = L & 15;
  const int n0 = w * 32, n1 = n0 + 16;   // each wave owns two 16-wide N tiles

  for (int t = 1; t < Sc; ++t) {
    // stage emissions for this step (coalesced, overlaps with GEMM issue)
    for (int bb = 0; bb < FB; ++bb)
      emit_sh[bb * Tc + tid] = logits[((size_t)(b0 + bb) * Sc + t) * Tc + tid];

    // (16 x 256) @ (256 x 256) f16 GEMM, f32 accumulate
    v8f acc0 = {0.f,0.f,0.f,0.f,0.f,0.f,0.f,0.f};
    v8f acc1 = {0.f,0.f,0.f,0.f,0.f,0.f,0.f,0.f};
#pragma unroll
    for (int kb = 0; kb < 8; ++kb) {
      const int k0 = kb * 32;
      // A fragment: elements e -> K=k0+16h+e (contiguous in aexp row)
      v8h alo = *(const v8h*)&aexp[mrow * AP + k0 + 16 * hh];
      v8h ahi = *(const v8h*)&aexp[mrow * AP + k0 + 16 * hh + 8];
      v16h af;
#pragma unroll
      for (int e = 0; e < 8; ++e) { af[e] = alo[e]; af[e + 8] = ahi[e]; }
      // B fragment: elements e -> K=k0+16h+e (contiguous in bT row)
      v8h b0lo = *(const v8h*)&bT[(n0 + ncol) * BP + k0 + 16 * hh];
      v8h b0hi = *(const v8h*)&bT[(n0 + ncol) * BP + k0 + 16 * hh + 8];
      v8h b1lo = *(const v8h*)&bT[(n1 + ncol) * BP + k0 + 16 * hh];
      v8h b1hi = *(const v8h*)&bT[(n1 + ncol) * BP + k0 + 16 * hh + 8];
      v16h bf0, bf1;
#pragma unroll
      for (int e = 0; e < 8; ++e) { bf0[e] = b0lo[e]; bf0[e + 8] = b0hi[e];
                                    bf1[e] = b1lo[e]; bf1[e + 8] = b1hi[e]; }
      acc0 = __builtin_amdgcn_wmma_f32_16x16x32_f16(false, af, false, bf0,
                                                    (short)0, acc0, false, false);
      acc1 = __builtin_amdgcn_wmma_f32_16x16x32_f16(false, af, false, bf1,
                                                    (short)0, acc1, false, false);
    }
    __syncthreads();   // emit_sh ready; all aexp reads complete

    // masked score update: D layout lane(n=L&15,h), vgpr v -> (m=v+8h, n)
#pragma unroll
    for (int v = 0; v < 8; ++v) {
      int bb = v + 8 * hh;
      if (t < len_sh[bb]) {
        int j0 = n0 + ncol;
        score[bb * Tc + j0] = mmax[bb] + __logf(acc0[v]) + emit_sh[bb * Tc + j0];
        int j1 = n1 + ncol;
        score[bb * Tc + j1] = mmax[bb] + __logf(acc1[v]) + emit_sh[bb * Tc + j1];
      }
    }
    __syncthreads();
    recompute();
  }

  // den[b] = logsumexp_j(score[b][j] + end[j])
  { int rb = tid >> 4, rl = tid & 15;
    float m = -3.0e38f;
    for (int s = 0; s < 16; ++s) { int j = rl + 16 * s;
      m = fmaxf(m, score[rb * Tc + j] + end_sh[j]); }
    red[tid] = m; __syncthreads();
    if (tid < FB) { float mm = -3.0e38f;
      for (int s = 0; s < 16; ++s) mm = fmaxf(mm, red[tid * 16 + s]);
      mmax[tid] = mm; }
    __syncthreads();
    float sum = 0.f;
    for (int s = 0; s < 16; ++s) { int j = rl + 16 * s;
      sum += __expf(score[rb * Tc + j] + end_sh[j] - mmax[rb]); }
    red[tid] = sum; __syncthreads();
    if (tid < FB) { float tt = 0.f;
      for (int s = 0; s < 16; ++s) tt += red[tid * 16 + s];
      den[b0 + tid] = mmax[tid] + __logf(tt); }
  }
}

// ---------------------------------------------------------------------------
// Viterbi (max-plus) — one block per batch, thread j owns next-tag j.
// trans staged in LDS so lane j reads trans[i*256+j] conflict-free.
// Back-pointers stored as 1 byte (T=256).
// ---------------------------------------------------------------------------
__global__ void crf_viterbi_kernel(const float* __restrict__ logits,
                                   const unsigned char* __restrict__ mask,
                                   const float* __restrict__ start_t,
                                   const float* __restrict__ end_t,
                                   const float* __restrict__ trans,
                                   unsigned char* __restrict__ hist_g,
                                   int* __restrict__ last_g) {
  extern __shared__ char smem[];
  float* trans_sh = (float*)smem;       // Tc*Tc floats (256 KB)
  __shared__ float score_sh[Tc];
  __shared__ float redf[Tc];
  __shared__ int   redi[Tc];
  __shared__ int   len_shv;

  const int j = threadIdx.x;
  const int b = blockIdx.x;

  for (int idx = j; idx < Tc * Tc; idx += Tc) trans_sh[idx] = trans[idx];

  { int c = 0; for (int t = j; t < Sc; t += Tc) c += (int)mask[(size_t)b * Sc + t];
    redi[j] = c; }
  __syncthreads();
  for (int s = 128; s > 0; s >>= 1) {
    if (j < s) redi[j] += redi[j + s];
    __syncthreads();
  }
  if (j == 0) len_shv = redi[0];

  score_sh[j] = start_t[j] + logits[((size_t)b * Sc) * Tc + j];
  __syncthreads();
  const int len = len_shv;

  for (int t = 1; t < Sc; ++t) {
    float emit = logits[((size_t)b * Sc + t) * Tc + j];
    float best = -3.0e38f; int bi = 0;
#pragma unroll 4
    for (int i = 0; i < Tc; ++i) {
      float cand = score_sh[i] + trans_sh[(i << 8) + j];
      bi   = (cand > best) ? i : bi;      // first-max tie-break like argmax
      best = fmaxf(cand, best);
    }
    __syncthreads();
    unsigned char h;
    if (t < len) { score_sh[j] = best + emit; h = (unsigned char)bi; }
    else          h = (unsigned char)j;     // identity history when masked
    hist_g[((size_t)b * Sc + t) * Tc + j] = h;
    __syncthreads();
  }

  // last = argmax_j(score + end)
  redf[j] = score_sh[j] + end_t[j]; redi[j] = j;
  __syncthreads();
  for (int s = 128; s > 0; s >>= 1) {
    if (j < s) {
      if (redf[j + s] > redf[j]) { redf[j] = redf[j + s]; redi[j] = redi[j + s]; }
    }
    __syncthreads();
  }
  if (j == 0) last_g[b] = redi[0];
}

// ---------------------------------------------------------------------------
// Gold-path (numerator) score — one block per batch.
// ---------------------------------------------------------------------------
__global__ void crf_numerator_kernel(const float* __restrict__ logits,
                                     const unsigned char* __restrict__ mask,
                                     const int* __restrict__ labels,
                                     const float* __restrict__ start_t,
                                     const float* __restrict__ end_t,
                                     const float* __restrict__ trans,
                                     float* __restrict__ num) {
  __shared__ float redf[256];
  __shared__ int   redi[256];
  __shared__ int   len_shn;
  const int tid = threadIdx.x, b = blockIdx.x;

  { int c = 0; for (int t = tid; t < Sc; t += 256) c += (int)mask[(size_t)b * Sc + t];
    redi[tid] = c; }
  __syncthreads();
  for (int s = 128; s > 0; s >>= 1) { if (tid < s) redi[tid] += redi[tid + s]; __syncthreads(); }
  if (tid == 0) len_shn = redi[0];
  __syncthreads();
  const int len = len_shn;

  float sc = 0.f;
  for (int t = tid; t < Sc; t += 256) {
    int lab = labels[(size_t)b * Sc + t];
    if (t == 0) {
      sc += start_t[lab] + logits[((size_t)b * Sc) * Tc + lab];
    } else if (t < len) {
      int prev = labels[(size_t)b * Sc + t - 1];
      sc += trans[prev * Tc + lab] + logits[((size_t)b * Sc + t) * Tc + lab];
    }
  }
  redf[tid] = sc; __syncthreads();
  for (int s = 128; s > 0; s >>= 1) { if (tid < s) redf[tid] += redf[tid + s]; __syncthreads(); }
  if (tid == 0) num[b] = redf[0] + end_t[labels[(size_t)b * Sc + len - 1]];
}

// ---------------------------------------------------------------------------
// Backtrace — stage the batch's 128 KB history in LDS, sequential walk in LDS.
// ---------------------------------------------------------------------------
__global__ void crf_backtrace_kernel(const unsigned char* __restrict__ hist_g,
                                     const unsigned char* __restrict__ mask,
                                     const int* __restrict__ last_g,
                                     float* __restrict__ pred_out) {
  extern __shared__ char smem[];
  unsigned char* hist_sh = (unsigned char*)smem;   // Sc*Tc = 128 KB
  __shared__ int pred_sh[Sc];
  __shared__ int redi[256];
  __shared__ int len_shb;
  const int tid = threadIdx.x, b = blockIdx.x;

  const uint4* src = (const uint4*)(hist_g + (size_t)b * Sc * Tc);
  uint4* dst = (uint4*)hist_sh;
  for (int idx = tid; idx < (Sc * Tc) / 16; idx += 256) dst[idx] = src[idx];

  { int c = 0; for (int t = tid; t < Sc; t += 256) c += (int)mask[(size_t)b * Sc + t];
    redi[tid] = c; }
  __syncthreads();
  for (int s = 128; s > 0; s >>= 1) { if (tid < s) redi[tid] += redi[tid + s]; __syncthreads(); }

  if (tid == 0) {
    len_shb = redi[0];
    int tag = last_g[b];
    for (int t = Sc - 1; t >= 1; --t) {
      pred_sh[t] = tag;
      tag = (int)hist_sh[t * Tc + tag];
    }
    pred_sh[0] = tag;
  }
  __syncthreads();
  const int len = len_shb;
  for (int t = tid; t < Sc; t += 256)
    pred_out[(size_t)b * Sc + t] = (t < len) ? (float)pred_sh[t] : 0.0f;
}

// ---------------------------------------------------------------------------
// loss = -mean(num - den)
// ---------------------------------------------------------------------------
__global__ void crf_loss_kernel(const float* __restrict__ num,
                                const float* __restrict__ den,
                                float* __restrict__ out_loss) {
  __shared__ float redf[128];
  const int tid = threadIdx.x;
  redf[tid] = num[tid] - den[tid];
  __syncthreads();
  for (int s = 64; s > 0; s >>= 1) { if (tid < s) redf[tid] += redf[tid + s]; __syncthreads(); }
  if (tid == 0) *out_loss = -redf[0] / (float)Bc;
}

extern "C" void kernel_launch(void* const* d_in, const int* in_sizes, int n_in,
                              void* d_out, int out_size, void* d_ws, size_t ws_size,
                              hipStream_t stream) {
  const float*         logits  = (const float*)d_in[0];
  const unsigned char* mask    = (const unsigned char*)d_in[1];  // jnp bool_ = 1 byte
  const int*           labels  = (const int*)d_in[2];
  const float*         start_t = (const float*)d_in[3];
  const float*         end_t   = (const float*)d_in[4];
  const float*         trans   = (const float*)d_in[5];

  // workspace layout
  unsigned char* hist = (unsigned char*)d_ws;                       // B*S*T = 16 MB
  float*         den  = (float*)((char*)d_ws + (size_t)Bc * Sc * Tc);
  float*         num  = den + Bc;
  int*           last = (int*)(num + Bc);

  float* pred = (float*)d_out;                   // B*S floats
  float* loss = pred + (size_t)Bc * Sc;          // 1 float

  const size_t vit_smem = (size_t)Tc * Tc * sizeof(float);                 // 256 KB
  const size_t fwd_smem = (size_t)(FB * Tc * 4) * 2 + Tc * 4 + 256 * 4 + FB * 4 + FB * 4
                        + (size_t)FB * AP * 2 + (size_t)Tc * BP * 2;       // ~175 KB
  const size_t bt_smem  = (size_t)Sc * Tc;                                 // 128 KB

  crf_viterbi_kernel<<<dim3(Bc), dim3(256), vit_smem, stream>>>(
      logits, mask, start_t, end_t, trans, hist, last);
  crf_forward_kernel<<<dim3(Bc / FB), dim3(256), fwd_smem, stream>>>(
      logits, mask, start_t, end_t, trans, den);
  crf_numerator_kernel<<<dim3(Bc), dim3(256), 0, stream>>>(
      logits, mask, labels, start_t, end_t, trans, num);
  crf_backtrace_kernel<<<dim3(Bc), dim3(256), bt_smem, stream>>>(
      hist, mask, last, pred);
  crf_loss_kernel<<<dim3(1), dim3(128), 0, stream>>>(num, den, loss);
}
